// MoE_90950227460276
// MI455X (gfx1250) — matrix operand order repaired and checked by reading, not compiled, not run on hardware
//
#include <hip/hip_runtime.h>
#include <math.h>

// ---------------- Types for CDNA5 WMMA ----------------
typedef __attribute__((ext_vector_type(16))) __bf16 v16bf;
typedef __attribute__((ext_vector_type(8)))  float  v8f;

union FragBF { uint4 u[2]; v16bf v; };

__device__ __forceinline__ unsigned short f2bf(float f) {
    unsigned u = __float_as_uint(f);
    unsigned r = u + 0x7FFFu + ((u >> 16) & 1u);   // RNE
    return (unsigned short)(r >> 16);
}
__device__ __forceinline__ unsigned pack2(unsigned short lo, unsigned short hi) {
    return (unsigned)lo | ((unsigned)hi << 16);
}

// Async global->LDS copy of 32 bytes (two b128 beats), tracked on ASYNCcnt.
// IOFFSET applies to both the LDS and the global address (ISA 08_async_tensor §4.4).
__device__ __forceinline__ void async_copy_b32bytes(void* lds_ptr, const void* gptr) {
    unsigned lds = (unsigned)(unsigned long long)lds_ptr;          // LDS offset = addr[31:0]
    unsigned long long g = (unsigned long long)gptr;
    asm volatile("global_load_async_to_lds_b128 %0, %1, off\n\t"
                 "global_load_async_to_lds_b128 %0, %1, off offset:16"
                 :: "v"(lds), "v"(g) : "memory");
}
__device__ __forceinline__ void async_wait0() {
    asm volatile("s_wait_asynccnt 0x0" ::: "memory");
}

#define NTOK 512
#define DIM 2048
#define NEXP 64
#define TOPK 6
#define CAP 128
#define MOE_I 1408
#define SH_I 2816

// ---------------- 1) x -> bf16 (with zero sentinel row at index 512) ----------------
__global__ void moe_cvt_x(const float* __restrict__ x, unsigned short* __restrict__ xbf) {
    int i = blockIdx.x * 256 + threadIdx.x;
    if (i >= (NTOK + 1) * DIM) return;
    float v = (i < NTOK * DIM) ? x[i] : 0.0f;
    xbf[i] = f2bf(v);
}

// ---------------- 2) gate: logits -> softmax -> top-6 ----------------
__global__ void moe_gate(const float* __restrict__ x, const float* __restrict__ gw,
                         float* __restrict__ gate_vals, int* __restrict__ gate_idx) {
    int t = blockIdx.x;
    int e = threadIdx.x;                 // 0..63
    const float* xr = x + (long)t * DIM;
    const float* wr = gw + (long)e * DIM;
    float acc = 0.f;
    for (int d = 0; d < DIM; d += 4) {
        float4 xv = *(const float4*)(xr + d);
        float4 wv = *(const float4*)(wr + d);
        acc += xv.x * wv.x + xv.y * wv.y + xv.z * wv.z + xv.w * wv.w;
    }
    __shared__ float sc[NEXP];
    sc[e] = acc;
    __syncthreads();
    if (e == 0) {
        float mx = sc[0];
        for (int i = 1; i < NEXP; i++) mx = sc[i] > mx ? sc[i] : mx;
        float s = 0.f;
        for (int i = 0; i < NEXP; i++) { float p = expf(sc[i] - mx); sc[i] = p; s += p; }
        float inv = 1.0f / s;
        for (int i = 0; i < NEXP; i++) sc[i] *= inv;
        for (int k = 0; k < TOPK; k++) {
            int bi = 0; float bv = sc[0];
            for (int i = 1; i < NEXP; i++) if (sc[i] > bv) { bv = sc[i]; bi = i; }
            gate_vals[t * TOPK + k] = bv;   // ROUTE_SCALE == 1.0
            gate_idx[t * TOPK + k] = bi;
            sc[bi] = -1.0f;
        }
    }
}

// ---------------- 3) dispatch: deterministic cumsum-order slot assignment ----------------
__global__ void moe_dispatch(const int* __restrict__ gate_idx,
                             int* __restrict__ buf, int* __restrict__ slot) {
    __shared__ int sidx[NTOK * TOPK];
    for (int i = threadIdx.x; i < NTOK * TOPK; i += 64) sidx[i] = gate_idx[i];
    __syncthreads();
    int e = threadIdx.x;                 // one expert per thread
    int cnt = 0;
    for (int p = 0; p < NTOK * TOPK; p++) {
        if (sidx[p] == e) {
            if (cnt < CAP) buf[e * CAP + cnt] = p / TOPK;
            slot[p] = cnt;
            cnt++;
        }
    }
    for (int c = cnt; c < CAP; c++) buf[e * CAP + c] = NTOK;   // sentinel -> zero row
}

// ---------------- 4) GEMM1: H = silu(A@W1^T) * (A@W3^T), bf16 out ----------------
// Block tile 128(M) x 64(N), K-step 32. 8 waves: 4(M) x 2(N), each wave 32x32.
__global__ __launch_bounds__(256) void moe_gemm1(
    const unsigned short* __restrict__ Abf,   // bf16 [*, K]
    const int* __restrict__ rowidx,           // per-expert [CAP] token ids, or null
    const float* __restrict__ W1, const float* __restrict__ W3, // f32 [N, K]
    unsigned short* __restrict__ Hout,        // bf16
    int K, int Nld, long wstride, long hstride)
{
    int e = blockIdx.z;
    int nbase = blockIdx.x * 64;
    int rowbase = blockIdx.y * 128;
    const float* w1 = W1 + (long)e * wstride;
    const float* w3 = W3 + (long)e * wstride;

    __shared__ unsigned short As[128 * 32];
    __shared__ unsigned short Bs1[64 * 32];
    __shared__ unsigned short Bs3[64 * 32];

    int tid = threadIdx.x, lane = tid & 31, wid = tid >> 5;
    int wm = (wid & 3) * 32, wn = (wid >> 2) * 32;

    // cooperative A: 2 threads per row, 16 halves each
    int ar = tid >> 1, aseg = (tid & 1) * 16;
    long arow = rowidx ? (long)rowidx[e * CAP + rowbase + ar] : (long)(rowbase + ar);
    const unsigned short* aptr = Abf + arow * (long)K;
    // cooperative B: 4 threads per row, 8 floats each
    int brow = tid >> 2, bk = (tid & 3) * 8;
    const float* b1p = w1 + (long)(nbase + brow) * K + bk;
    const float* b3p = w3 + (long)(nbase + brow) * K + bk;

    v8f acc1[2][2] = {};
    v8f acc3[2][2] = {};

    int rsel = lane & 15;
    int hs   = (lane >> 4) * 8;    // A: half-selection offset (halves)
    int koff = (lane >> 4) * 8;    // B: K offset (halves)

    int ksteps = K >> 5;
    for (int kk = 0; kk < ksteps; kk++) {
        int k0 = kk * 32;
        // A tile (already bf16): async global -> LDS, no VGPR round-trip
        async_copy_b32bytes(&As[ar * 32 + aseg], aptr + k0 + aseg);
        { // B tiles: f32 -> bf16 in flight
            float4 f0 = *(const float4*)(b1p + k0);
            float4 f1 = *(const float4*)(b1p + k0 + 4);
            uint4 p;
            p.x = pack2(f2bf(f0.x), f2bf(f0.y)); p.y = pack2(f2bf(f0.z), f2bf(f0.w));
            p.z = pack2(f2bf(f1.x), f2bf(f1.y)); p.w = pack2(f2bf(f1.z), f2bf(f1.w));
            *(uint4*)(&Bs1[brow * 32 + bk]) = p;
            f0 = *(const float4*)(b3p + k0);
            f1 = *(const float4*)(b3p + k0 + 4);
            p.x = pack2(f2bf(f0.x), f2bf(f0.y)); p.y = pack2(f2bf(f0.z), f2bf(f0.w));
            p.z = pack2(f2bf(f1.x), f2bf(f1.y)); p.w = pack2(f2bf(f1.z), f2bf(f1.w));
            *(uint4*)(&Bs3[brow * 32 + bk]) = p;
            __builtin_prefetch(b1p + k0 + 32, 0, 0);   // global_prefetch_b8
            __builtin_prefetch(b3p + k0 + 32, 0, 0);
        }
        async_wait0();                 // A tile landed in LDS
        __syncthreads();

        FragBF a[2], b1[2], b3[2];
        #pragma unroll
        for (int mi = 0; mi < 2; mi++) {
            int r = wm + mi * 16 + rsel;
            a[mi].u[0] = *(const uint4*)(&As[r * 32 + hs]);
            a[mi].u[1] = *(const uint4*)(&As[r * 32 + 16 + hs]);
        }
        #pragma unroll
        for (int ni = 0; ni < 2; ni++) {
            int n = wn + ni * 16 + rsel;
            b1[ni].u[0] = *(const uint4*)(&Bs1[n * 32 + koff]);
            b1[ni].u[1] = *(const uint4*)(&Bs1[n * 32 + koff + 16]);
            b3[ni].u[0] = *(const uint4*)(&Bs3[n * 32 + koff]);
            b3[ni].u[1] = *(const uint4*)(&Bs3[n * 32 + koff + 16]);
        }
        #pragma unroll
        for (int mi = 0; mi < 2; mi++)
            #pragma unroll
            for (int ni = 0; ni < 2; ni++) {
                acc1[mi][ni] = __builtin_amdgcn_wmma_f32_16x16x32_bf16(
                    false, a[mi].v, false, b1[ni].v, (short)0, acc1[mi][ni], false, false);
                acc3[mi][ni] = __builtin_amdgcn_wmma_f32_16x16x32_bf16(
                    false, a[mi].v, false, b3[ni].v, (short)0, acc3[mi][ni], false, false);
            }
        __syncthreads();
    }

    // epilogue: H = silu(g1) * g3, bf16
    int moff = 8 * (lane >> 4);
    int ncol = lane & 15;
    #pragma unroll
    for (int mi = 0; mi < 2; mi++)
        #pragma unroll
        for (int ni = 0; ni < 2; ni++)
            #pragma unroll
            for (int j = 0; j < 8; j++) {
                int m = wm + mi * 16 + j + moff;
                int n = wn + ni * 16 + ncol;
                float g = acc1[mi][ni][j];
                float u = acc3[mi][ni][j];
                float hv = (g / (1.0f + expf(-g))) * u;
                Hout[e * hstride + (long)(rowbase + m) * Nld + (nbase + n)] = f2bf(hv);
            }
}

// ---------------- 5) GEMM2: Y = H @ W2^T, f32 out ----------------
__global__ __launch_bounds__(256) void moe_gemm2(
    const unsigned short* __restrict__ Abf,   // bf16 [M, K] (per-expert offset via astride)
    const float* __restrict__ W2,             // f32 [N, K]
    float* __restrict__ Yout,
    int K, int Nld, long astride, long wstride, long ystride)
{
    int e = blockIdx.z;
    int nbase = blockIdx.x * 64;
    int rowbase = blockIdx.y * 128;
    const float* w2 = W2 + (long)e * wstride;
    const unsigned short* Ab = Abf + (long)e * astride;

    __shared__ unsigned short As[128 * 32];
    __shared__ unsigned short Bs[64 * 32];

    int tid = threadIdx.x, lane = tid & 31, wid = tid >> 5;
    int wm = (wid & 3) * 32, wn = (wid >> 2) * 32;

    int ar = tid >> 1, aseg = (tid & 1) * 16;
    const unsigned short* aptr = Ab + (long)(rowbase + ar) * K;
    int brow = tid >> 2, bk = (tid & 3) * 8;
    const float* bp = w2 + (long)(nbase + brow) * K + bk;

    v8f acc[2][2] = {};
    int rsel = lane & 15;
    int hs   = (lane >> 4) * 8;
    int koff = (lane >> 4) * 8;

    int ksteps = K >> 5;
    for (int kk = 0; kk < ksteps; kk++) {
        int k0 = kk * 32;
        // A tile async global -> LDS
        async_copy_b32bytes(&As[ar * 32 + aseg], aptr + k0 + aseg);
        {
            float4 f0 = *(const float4*)(bp + k0);
            float4 f1 = *(const float4*)(bp + k0 + 4);
            uint4 p;
            p.x = pack2(f2bf(f0.x), f2bf(f0.y)); p.y = pack2(f2bf(f0.z), f2bf(f0.w));
            p.z = pack2(f2bf(f1.x), f2bf(f1.y)); p.w = pack2(f2bf(f1.z), f2bf(f1.w));
            *(uint4*)(&Bs[brow * 32 + bk]) = p;
            __builtin_prefetch(bp + k0 + 32, 0, 0);
        }
        async_wait0();
        __syncthreads();

        FragBF a[2], b[2];
        #pragma unroll
        for (int mi = 0; mi < 2; mi++) {
            int r = wm + mi * 16 + rsel;
            a[mi].u[0] = *(const uint4*)(&As[r * 32 + hs]);
            a[mi].u[1] = *(const uint4*)(&As[r * 32 + 16 + hs]);
        }
        #pragma unroll
        for (int ni = 0; ni < 2; ni++) {
            int n = wn + ni * 16 + rsel;
            b[ni].u[0] = *(const uint4*)(&Bs[n * 32 + koff]);
            b[ni].u[1] = *(const uint4*)(&Bs[n * 32 + koff + 16]);
        }
        #pragma unroll
        for (int mi = 0; mi < 2; mi++)
            #pragma unroll
            for (int ni = 0; ni < 2; ni++)
                acc[mi][ni] = __builtin_amdgcn_wmma_f32_16x16x32_bf16(
                    false, a[mi].v, false, b[ni].v, (short)0, acc[mi][ni], false, false);
        __syncthreads();
    }

    int moff = 8 * (lane >> 4);
    int ncol = lane & 15;
    #pragma unroll
    for (int mi = 0; mi < 2; mi++)
        #pragma unroll
        for (int ni = 0; ni < 2; ni++)
            #pragma unroll
            for (int j = 0; j < 8; j++) {
                int m = wm + mi * 16 + j + moff;
                int n = wn + ni * 16 + ncol;
                Yout[e * ystride + (long)(rowbase + m) * Nld + (nbase + n)] = acc[mi][ni][j];
            }
}

// ---------------- 6) combine: out = zs + sum_k gate * ye[e, slot] ----------------
__global__ void moe_combine(const float* __restrict__ zs, const float* __restrict__ ye,
                            const float* __restrict__ gate_vals, const int* __restrict__ gate_idx,
                            const int* __restrict__ slot, float* __restrict__ out) {
    int t = blockIdx.x;
    int d0 = threadIdx.x * 8;
    float r[8];
    #pragma unroll
    for (int j = 0; j < 8; j++) r[j] = zs[(long)t * DIM + d0 + j];
    for (int k = 0; k < TOPK; k++) {
        int p = t * TOPK + k;
        int s = slot[p];
        if (s < CAP) {
            int e = gate_idx[p];
            float g = gate_vals[p];
            const float* yr = ye + ((long)e * CAP + s) * DIM + d0;
            #pragma unroll
            for (int j = 0; j < 8; j++) r[j] += g * yr[j];
        }
    }
    #pragma unroll
    for (int j = 0; j < 8; j++) out[(long)t * DIM + d0 + j] = r[j];
}

// ---------------- host ----------------
extern "C" void kernel_launch(void* const* d_in, const int* in_sizes, int n_in,
                              void* d_out, int out_size, void* d_ws, size_t ws_size,
                              hipStream_t stream) {
    const float* x      = (const float*)d_in[0];
    const float* gate_w = (const float*)d_in[1];
    const float* w1     = (const float*)d_in[2];
    const float* w2     = (const float*)d_in[3];
    const float* w3     = (const float*)d_in[4];
    const float* sw1    = (const float*)d_in[5];
    const float* sw2    = (const float*)d_in[6];
    const float* sw3    = (const float*)d_in[7];
    float* out = (float*)d_out;

    char* w = (char*)d_ws;
    size_t off = 0;
    auto alloc = [&](size_t bytes) { void* p = w + off; off = (off + bytes + 255) & ~(size_t)255; return p; };
    unsigned short* xbf  = (unsigned short*)alloc((size_t)(NTOK + 1) * DIM * 2);
    float* gate_vals     = (float*)alloc((size_t)NTOK * TOPK * 4);
    int*   gate_idx      = (int*)  alloc((size_t)NTOK * TOPK * 4);
    int*   buf           = (int*)  alloc((size_t)NEXP * CAP * 4);
    int*   slot          = (int*)  alloc((size_t)NTOK * TOPK * 4);
    unsigned short* hR   = (unsigned short*)alloc((size_t)NEXP * CAP * MOE_I * 2);
    unsigned short* hS   = (unsigned short*)alloc((size_t)NTOK * SH_I * 2);
    float* ye            = (float*)alloc((size_t)NEXP * CAP * DIM * 4);
    float* zs            = (float*)alloc((size_t)NTOK * DIM * 4);

    moe_cvt_x<<<((NTOK + 1) * DIM + 255) / 256, 256, 0, stream>>>(x, xbf);
    moe_gate<<<NTOK, 64, 0, stream>>>(x, gate_w, gate_vals, gate_idx);
    moe_dispatch<<<1, 64, 0, stream>>>(gate_idx, buf, slot);

    // routed GEMM1: per expert [128 x 2048] x [2048 x 1408] (w1 & w3, fused SwiGLU)
    moe_gemm1<<<dim3(MOE_I / 64, 1, NEXP), 256, 0, stream>>>(
        xbf, buf, w1, w3, hR, DIM, MOE_I, (long)MOE_I * DIM, (long)CAP * MOE_I);
    // shared GEMM1: [512 x 2048] x [2048 x 2816]
    moe_gemm1<<<dim3(SH_I / 64, NTOK / 128, 1), 256, 0, stream>>>(
        xbf, nullptr, sw1, sw3, hS, DIM, SH_I, 0, 0);
    // routed GEMM2: per expert [128 x 1408] x [1408 x 2048]
    moe_gemm2<<<dim3(DIM / 64, 1, NEXP), 256, 0, stream>>>(
        hR, w2, ye, MOE_I, DIM, (long)CAP * MOE_I, (long)DIM * MOE_I, (long)CAP * DIM);
    // shared GEMM2: [512 x 2816] x [2816 x 2048]
    moe_gemm2<<<dim3(DIM / 64, NTOK / 128, 1), 256, 0, stream>>>(
        hS, sw2, zs, SH_I, DIM, 0, 0, 0);

    moe_combine<<<NTOK, 256, 0, stream>>>(zs, ye, gate_vals, gate_idx, slot, out);
}
